// GatherModel_81011673137217
// MI455X (gfx1250) — matrix operand order repaired and checked by reading, not compiled, not run on hardware
//
#include <hip/hip_runtime.h>

#define NLIG 5000
#define NREC 15000
#define ELIG 50000
#define EREC 150000
#define NSTEPS 6

typedef __attribute__((ext_vector_type(16))) _Float16 v16h;
typedef __attribute__((ext_vector_type(8)))  float    v8f;
typedef __attribute__((ext_vector_type(4)))  float    f4;
typedef __attribute__((ext_vector_type(8)))  _Float16 h8;

// ---------------------------------------------------------------------------
// WMMA 16x16x32 f16 operand loaders (CDNA5 ISA 7.12.2, wave32).
// A (16x32 MxK): lane m=l&15, hi=l>>4. Half t = A[row][k0+16*hi+t]
//                                      (matches B's k order)
//   -> two contiguous 8-element runs per lane: vectorizable.
// B (32x16 KxN): lane n=l&15, hi=l>>4. Half t = B[k0+16*hi+t][col].
//   With torch layout W[col*ldw+k] this is 16 contiguous floats: vectorizable.
// C/D (16x16 f32): lane n=l&15, hi=l>>4; VGPR r = row r+8*hi.
// ---------------------------------------------------------------------------
__device__ inline v16h load_a_fast(const float* __restrict__ Ar, int k0, int hi) {
  const float* p0 = Ar + k0 + 16 * hi;
  const float* p1 = Ar + k0 + 16 * hi + 8;
  f4 x0 = *(const f4*)(p0);
  f4 x1 = *(const f4*)(p0 + 4);
  f4 x2 = *(const f4*)(p1);
  f4 x3 = *(const f4*)(p1 + 4);
  v16h a;
#pragma unroll
  for (int i = 0; i < 4; ++i) {
    a[i]      = (_Float16)x0[i];
    a[4 + i]  = (_Float16)x1[i];
    a[8 + i]  = (_Float16)x2[i];
    a[12 + i] = (_Float16)x3[i];
  }
  return a;
}
__device__ inline v16h load_a_fast(const _Float16* __restrict__ Ar, int k0, int hi) {
  h8 x0 = *(const h8*)(Ar + k0 + 16 * hi);
  h8 x1 = *(const h8*)(Ar + k0 + 16 * hi + 8);
  v16h a;
#pragma unroll
  for (int i = 0; i < 8; ++i) { a[i] = x0[i]; a[8 + i] = x1[i]; }
  return a;
}

// Guarded (edge-tile) A loader: clamped addresses + selects, no EXEC branches.
template<typename AT>
__device__ inline v16h load_a_guard(const AT* __restrict__ A, int N, int K,
                                    int row, int k0, int hi) {
  int rc = row < N ? row : (N - 1);
  bool rok = row < N;
  const AT* Ar = A + (size_t)rc * (size_t)K;
  v16h a;
#pragma unroll
  for (int j = 0; j < 8; ++j) {
    int kb = k0 + 16 * hi + 2 * j;
    int kc0 = kb < K ? kb : K - 1;
    int kc1 = kb + 1 < K ? kb + 1 : K - 1;
    float v0 = (float)Ar[kc0];
    float v1 = (float)Ar[kc1];
    a[2 * j]     = (rok && kb     < K) ? (_Float16)v0 : (_Float16)0.f;
    a[2 * j + 1] = (rok && kb + 1 < K) ? (_Float16)v1 : (_Float16)0.f;
  }
  return a;
}

// B loaders; W is torch layout: B[k][o] = W[o*ldw + k].
__device__ inline v16h load_b_fast(const float* __restrict__ W, size_t ldw,
                                   int col, int k0, int hi) {
  const float* p = W + (size_t)col * ldw + k0 + 16 * hi;
  f4 x0 = *(const f4*)(p);
  f4 x1 = *(const f4*)(p + 4);
  f4 x2 = *(const f4*)(p + 8);
  f4 x3 = *(const f4*)(p + 12);
  v16h b;
#pragma unroll
  for (int i = 0; i < 4; ++i) {
    b[i]      = (_Float16)x0[i];
    b[4 + i]  = (_Float16)x1[i];
    b[8 + i]  = (_Float16)x2[i];
    b[12 + i] = (_Float16)x3[i];
  }
  return b;
}
__device__ inline v16h load_b_guard(const float* __restrict__ W, size_t ldw,
                                    int K, int O, int col, int k0, int hi) {
  int cc = col < O ? col : O - 1;
  bool cok = col < O;
  const float* p = W + (size_t)cc * ldw;
  v16h b;
#pragma unroll
  for (int t = 0; t < 16; ++t) {
    int k = k0 + 16 * hi + t;
    int kc = k < K ? k : K - 1;
    float v = p[kc];
    b[t] = (cok && k < K) ? (_Float16)v : (_Float16)0.f;
  }
  return b;
}

// ---------------------------------------------------------------------------
// C[NxO] = act(A[NxK] @ B + bias + Cadd). One wave computes a 16x64 strip
// (4 col tiles, A tile reused 4x). Requires O % 64 == 0.
// act: 0=none, 1=relu, 2=leaky-relu(0.01)
// ---------------------------------------------------------------------------
template<typename AT, typename OT>
__global__ __launch_bounds__(32) void linear_wmma(
    const AT* __restrict__ A, const float* __restrict__ W,
    const float* __restrict__ bias, const float* __restrict__ Cadd,
    OT* __restrict__ Cout, int N, int K, int O, int ldw, int act) {
  int row0 = blockIdx.x * 16, colBase = blockIdx.y * 64;
  int lane = threadIdx.x & 31;
  int m = lane & 15, hi = lane >> 4;
  bool rowsFull = (row0 + 15 < N);                 // uniform
  int row = row0 + m;
  int rc = row < N ? row : N - 1;
  const AT* Ar = A + (size_t)rc * (size_t)K;

  v8f zero = {};
  v8f acc[4] = {zero, zero, zero, zero};
  for (int k0 = 0; k0 < K; k0 += 32) {
    bool kfull = (k0 + 31 < K);                    // uniform
    v16h a = (rowsFull && kfull) ? load_a_fast(Ar, k0, hi)
                                 : load_a_guard<AT>(A, N, K, row, k0, hi);
#pragma unroll
    for (int c = 0; c < 4; ++c) {
      int col = colBase + 16 * c + m;
      v16h b = kfull ? load_b_fast(W, (size_t)ldw, col, k0, hi)
                     : load_b_guard(W, (size_t)ldw, K, O, col, k0, hi);
      acc[c] = __builtin_amdgcn_wmma_f32_16x16x32_f16(false, a, false, b,
                                                      (short)0, acc[c], false, false);
    }
  }
#pragma unroll
  for (int c = 0; c < 4; ++c) {
    int col = colBase + 16 * c + m;
    float bv = bias ? bias[col] : 0.0f;
#pragma unroll
    for (int r = 0; r < 8; ++r) {
      int rr = row0 + r + 8 * hi;
      if (rr < N) {
        float v = acc[c][r] + bv;
        if (Cadd) v += Cadd[(size_t)rr * O + col];
        if (act == 1) v = fmaxf(v, 0.f);
        else if (act == 2) v = (v > 0.f) ? v : 0.01f * v;
        Cout[(size_t)rr * O + col] = (OT)v;
      }
    }
  }
}

// ---------------------------------------------------------------------------
// Scores: P[i,j] = mask*(q_i.k_j) - 1000*(1-mask), f16 out. 16x64 per wave.
// Km is (Nk x 64) row-major == torch layout B with ldw=64.
// ---------------------------------------------------------------------------
__global__ __launch_bounds__(32) void score_wmma(
    const float* __restrict__ Q, const float* __restrict__ Km,
    const float* __restrict__ mask, size_t ms0, size_t ms1,
    _Float16* __restrict__ P, int Nq, int Nk) {
  int row0 = blockIdx.x * 16, colBase = blockIdx.y * 64;
  int lane = threadIdx.x & 31;
  int m = lane & 15, hi = lane >> 4;
  bool rowsFull = (row0 + 15 < Nq);
  int row = row0 + m;
  int rcl = row < Nq ? row : Nq - 1;
  const float* Ar = Q + (size_t)rcl * 64;

  v8f zero = {};
  v8f acc[4] = {zero, zero, zero, zero};
#pragma unroll
  for (int k0 = 0; k0 < 64; k0 += 32) {
    v16h a = rowsFull ? load_a_fast(Ar, k0, hi)
                      : load_a_guard<float>(Q, Nq, 64, row, k0, hi);
#pragma unroll
    for (int c = 0; c < 4; ++c) {
      int colT = colBase + 16 * c;
      v16h b = (colT + 15 < Nk) ? load_b_fast(Km, 64, colT + m, k0, hi)
                                : load_b_guard(Km, 64, 64, Nk, colT + m, k0, hi);
      acc[c] = __builtin_amdgcn_wmma_f32_16x16x32_f16(false, a, false, b,
                                                      (short)0, acc[c], false, false);
    }
  }
#pragma unroll
  for (int c = 0; c < 4; ++c) {
    int col = colBase + 16 * c + m;
    if (col >= Nk) continue;
#pragma unroll
    for (int r = 0; r < 8; ++r) {
      int rr = row0 + r + 8 * hi;
      if (rr < Nq) {
        float mv = mask[(size_t)rr * ms0 + (size_t)col * ms1];
        float s = mv * acc[c][r] - 1000.0f * (1.0f - mv);
        P[(size_t)rr * Nk + col] = (_Float16)s;
      }
    }
  }
}

// Row softmax in place on f16 matrix (fp32 math). One block per row.
__global__ __launch_bounds__(256) void softmax_rows(_Float16* __restrict__ P,
                                                    int Nrows, int Ncols) {
  int row = blockIdx.x;
  if (row >= Nrows) return;
  _Float16* p = P + (size_t)row * Ncols;
  __shared__ float red[256];
  int t = threadIdx.x;
  float mx = -3.0e38f;
  for (int j = t; j < Ncols; j += 256) mx = fmaxf(mx, (float)p[j]);
  red[t] = mx; __syncthreads();
  for (int s = 128; s > 0; s >>= 1) { if (t < s) red[t] = fmaxf(red[t], red[t + s]); __syncthreads(); }
  mx = red[0]; __syncthreads();
  float sum = 0.f;
  for (int j = t; j < Ncols; j += 256) sum += __expf((float)p[j] - mx);
  red[t] = sum; __syncthreads();
  for (int s = 128; s > 0; s >>= 1) { if (t < s) red[t] += red[t + s]; __syncthreads(); }
  float inv = 1.0f / red[0];
  for (int j = t; j < Ncols; j += 256) p[j] = (_Float16)(__expf((float)p[j] - mx) * inv);
}

// msg[e,o] = hb[src,o] + sum_k z[e,k]*G[src, o*64+k]; atomic scatter to agg[dst].
// 4 edges / 256-thread block; thread = output channel. G rows stream from L2.
__global__ __launch_bounds__(256) void edge_msg(
    const float* __restrict__ z, const _Float16* __restrict__ G,
    const float* __restrict__ hb, const int* __restrict__ src,
    const int* __restrict__ dst, float* __restrict__ agg, int E) {
  int el = threadIdx.x >> 6;
  int o  = threadIdx.x & 63;
  int e  = blockIdx.x * 4 + el;
  __shared__ float zs[4][64];
  if (e < E) zs[el][o] = z[(size_t)e * 64 + o];
  __syncthreads();
  if (e >= E) return;
  int s = src[e];
  const h8* Gp = (const h8*)(G + (size_t)s * 4096 + (size_t)o * 64);
  float acc = hb[(size_t)s * 64 + o];
#pragma unroll
  for (int q = 0; q < 8; ++q) {
    h8 gv = Gp[q];
#pragma unroll
    for (int i = 0; i < 8; ++i) acc += zs[el][q * 8 + i] * (float)gv[i];
  }
  atomicAdd(&agg[(size_t)dst[e] * 64 + o], acc);
}

// LayerNorm over dim 64, optional residual add before normalization.
__global__ __launch_bounds__(64) void layernorm64(
    const float* __restrict__ X, const float* __restrict__ Res,
    const float* __restrict__ g, const float* __restrict__ b,
    float* __restrict__ Y, int N) {
  int row = blockIdx.x;
  if (row >= N) return;
  int t = threadIdx.x;
  float x = X[(size_t)row * 64 + t];
  if (Res) x += Res[(size_t)row * 64 + t];
  __shared__ float red[64];
  red[t] = x; __syncthreads();
  for (int s = 32; s > 0; s >>= 1) { if (t < s) red[t] += red[t + s]; __syncthreads(); }
  float mu = red[0] * (1.0f / 64.0f); __syncthreads();
  float d = x - mu;
  red[t] = d * d; __syncthreads();
  for (int s = 32; s > 0; s >>= 1) { if (t < s) red[t] += red[t + s]; __syncthreads(); }
  float var = red[0] * (1.0f / 64.0f);
  Y[(size_t)row * 64 + t] = d * rsqrtf(var + 1e-5f) * g[t] + b[t];
}

// out[c*R + r] = in[r*C + c]
__global__ __launch_bounds__(256) void transpose_f(const float* __restrict__ in,
                                                   float* __restrict__ out, int R, int C) {
  int idx = blockIdx.x * 256 + threadIdx.x;
  if (idx < R * C) {
    int r = idx / C, c = idx - r * C;
    out[(size_t)c * R + r] = in[idx];
  }
}

// ------------------------------ host helpers ------------------------------
static inline void lin_ff(const float* A, const float* W, const float* bias,
                          const float* Cadd, float* C, int N, int K, int O, int ldw,
                          int act, hipStream_t s) {
  dim3 g((N + 15) / 16, O / 64);
  linear_wmma<float, float><<<g, dim3(32), 0, s>>>(A, W, bias, Cadd, C, N, K, O, ldw, act);
}
static inline void lin_fh(const float* A, const float* W, _Float16* C, int N, int K,
                          int O, int ldw, hipStream_t s) {
  dim3 g((N + 15) / 16, O / 64);
  linear_wmma<float, _Float16><<<g, dim3(32), 0, s>>>(A, W, nullptr, nullptr, C, N, K, O, ldw, 0);
}
static inline void lin_hf(const _Float16* A, const float* W, float* C, int N, int K,
                          int O, int ldw, hipStream_t s) {
  dim3 g((N + 15) / 16, O / 64);
  linear_wmma<_Float16, float><<<g, dim3(32), 0, s>>>(A, W, nullptr, nullptr, C, N, K, O, ldw, 0);
}

struct BranchBufs {
  float* z; _Float16* G; float* hA; float* hB;
  float* hb; float* agg; float* m; float* t1;
};

static const float* run_branch(const float* x, const float* ef, const int* src,
                               const int* dst, int N, int E,
                               const float* W0, const float* b0,
                               const float* We1, const float* be1,
                               const float* We2T, const float* be2T,
                               const float* resW, const float* convb,
                               const float* msgW, const float* msgb,
                               BranchBufs B, hipStream_t s) {
  // z = relu(ef @ We1.T + be1)                (E x 64)
  lin_ff(ef, We1, be1, nullptr, B.z, E, 16, 64, 16, 1, s);
  // h0 = relu(x @ W0.T + b0)                  (N x 64)
  lin_ff(x, W0, b0, nullptr, B.hA, N, 64, 64, 64, 1, s);
  float* hc = B.hA; float* hn = B.hB;
  for (int step = 0; step < NSTEPS; ++step) {
    // G[n, c] = sum_i h[n,i] * We2T[c*64 + i]   (N x 4096, f16, L2-resident)
    lin_fh(hc, We2T, B.G, N, 64, 4096, 64, s);
    // hb[n,o] = sum_i h[n,i] * be2T[o*64 + i]
    lin_ff(hc, be2T, nullptr, nullptr, B.hb, N, 64, 64, 64, 0, s);
    hipMemsetAsync(B.agg, 0, (size_t)N * 64 * sizeof(float), s);
    edge_msg<<<dim3((E + 3) / 4), dim3(256), 0, s>>>(B.z, B.G, B.hb, src, dst, B.agg, E);
    // m = relu(agg + h @ resW.T + conv_b)
    lin_ff(hc, resW, convb, B.agg, B.m, N, 64, 64, 64, 1, s);
    // hnew = m @ msgW[:, :64].T + h @ msgW[:, 64:].T + msgb
    lin_ff(B.m, msgW, nullptr, nullptr, B.t1, N, 64, 64, 128, 0, s);
    lin_ff(hc, msgW + 64, msgb, B.t1, hn, N, 64, 64, 128, 0, s);
    float* tmp = hc; hc = hn; hn = tmp;
  }
  return hc;   // final h (before +x residual)
}

extern "C" void kernel_launch(void* const* d_in, const int* in_sizes, int n_in,
                              void* d_out, int out_size, void* d_ws, size_t ws_size,
                              hipStream_t stream) {
  (void)in_sizes; (void)n_in; (void)out_size; (void)ws_size;
  const float* lig_x   = (const float*)d_in[0];
  const float* lig_ef  = (const float*)d_in[1];
  const int*   lig_src = (const int*)d_in[2];
  const int*   lig_dst = (const int*)d_in[3];
  const float* rec_x   = (const float*)d_in[4];
  const float* rec_ef  = (const float*)d_in[5];
  const int*   rec_src = (const int*)d_in[6];
  const int*   rec_dst = (const int*)d_in[7];
  const float* mask    = (const float*)d_in[8];
  const float* W0    = (const float*)d_in[9];
  const float* b0    = (const float*)d_in[10];
  const float* We1   = (const float*)d_in[11];
  const float* be1   = (const float*)d_in[12];
  const float* We2   = (const float*)d_in[13];
  const float* be2   = (const float*)d_in[14];
  const float* resW  = (const float*)d_in[15];
  const float* convb = (const float*)d_in[16];
  const float* msgW  = (const float*)d_in[17];
  const float* msgb  = (const float*)d_in[18];
  const float* Qlig  = (const float*)d_in[19];
  const float* Klig  = (const float*)d_in[20];
  const float* Vlig  = (const float*)d_in[21];
  const float* Qrec  = (const float*)d_in[22];
  const float* Krec  = (const float*)d_in[23];
  const float* Vrec  = (const float*)d_in[24];
  const float* g_lig = (const float*)d_in[25];
  const float* b_lig = (const float*)d_in[26];
  const float* g_rec = (const float*)d_in[27];
  const float* b_rec = (const float*)d_in[28];

  // --- carve workspace ---
  char* wsp = (char*)d_ws;
  size_t off = 0;
  auto carve = [&](size_t bytes) -> char* {
    char* p = wsp + off;
    off += (bytes + 255) & ~(size_t)255;
    return p;
  };
  _Float16* P  = (_Float16*)carve((size_t)NLIG * NREC * sizeof(_Float16)); // 150 MB
  _Float16* G  = (_Float16*)carve((size_t)NREC * 4096 * sizeof(_Float16)); // 123 MB (L2-resident)
  float* We2T = (float*)carve((size_t)4096 * 64 * 4);  // We2T[c*64+i] = We2[i*4096+c]
  float* be2T = (float*)carve((size_t)64 * 64 * 4);    // be2T[o*64+k] = be2[k*64+o]
  float* zL  = (float*)carve((size_t)ELIG * 64 * 4);
  float* zR  = (float*)carve((size_t)EREC * 64 * 4);
  float* hA  = (float*)carve((size_t)NREC * 64 * 4);
  float* hB  = (float*)carve((size_t)NREC * 64 * 4);
  float* hb  = (float*)carve((size_t)NREC * 64 * 4);
  float* agg = (float*)carve((size_t)NREC * 64 * 4);
  float* mb  = (float*)carve((size_t)NREC * 64 * 4);
  float* t1  = (float*)carve((size_t)NREC * 64 * 4);
  float* hl  = (float*)carve((size_t)NLIG * 64 * 4);
  float* hr  = (float*)carve((size_t)NREC * 64 * 4);
  float* ql  = (float*)carve((size_t)NLIG * 64 * 4);
  float* kl  = (float*)carve((size_t)NLIG * 64 * 4);
  float* vl  = (float*)carve((size_t)NLIG * 64 * 4);
  float* qr  = (float*)carve((size_t)NREC * 64 * 4);
  float* kr  = (float*)carve((size_t)NREC * 64 * 4);
  float* vr  = (float*)carve((size_t)NREC * 64 * 4);
  float* vlT = (float*)carve((size_t)64 * NLIG * 4);
  float* vrT = (float*)carve((size_t)64 * NREC * 4);
  float* attL = (float*)carve((size_t)NLIG * 64 * 4);
  float* attR = (float*)carve((size_t)NREC * 64 * 4);

  // --- one-time weight re-layouts (k-contiguous B operands) ---
  transpose_f<<<dim3((4096 * 64 + 255) / 256), dim3(256), 0, stream>>>(We2, We2T, 64, 4096);
  transpose_f<<<dim3((64 * 64 + 255) / 256), dim3(256), 0, stream>>>(be2, be2T, 64, 64);

  BranchBufs BB{nullptr, G, hA, hB, hb, agg, mb, t1};

  // --- ligand branch, then hl = LN(h + x) ---
  BB.z = zL;
  const float* hfL = run_branch(lig_x, lig_ef, lig_src, lig_dst, NLIG, ELIG,
                                W0, b0, We1, be1, We2T, be2T, resW, convb, msgW, msgb,
                                BB, stream);
  layernorm64<<<dim3(NLIG), dim3(64), 0, stream>>>(hfL, lig_x, g_lig, b_lig, hl, NLIG);

  // --- receptor branch, then hr = LN(h + x) ---
  BB.z = zR;
  const float* hfR = run_branch(rec_x, rec_ef, rec_src, rec_dst, NREC, EREC,
                                W0, b0, We1, be1, We2T, be2T, resW, convb, msgW, msgb,
                                BB, stream);
  layernorm64<<<dim3(NREC), dim3(64), 0, stream>>>(hfR, rec_x, g_rec, b_rec, hr, NREC);

  // --- attention projections (no bias) ---
  lin_ff(hl, Qlig, nullptr, nullptr, ql, NLIG, 64, 64, 64, 2, stream); // lrelu
  lin_ff(hl, Klig, nullptr, nullptr, kl, NLIG, 64, 64, 64, 2, stream); // lrelu
  lin_ff(hl, Vlig, nullptr, nullptr, vl, NLIG, 64, 64, 64, 0, stream);
  lin_ff(hr, Qrec, nullptr, nullptr, qr, NREC, 64, 64, 64, 2, stream); // lrelu
  lin_ff(hr, Krec, nullptr, nullptr, kr, NREC, 64, 64, 64, 2, stream); // lrelu
  lin_ff(hr, Vrec, nullptr, nullptr, vr, NREC, 64, 64, 64, 0, stream);
  transpose_f<<<dim3((NLIG * 64 + 255) / 256), dim3(256), 0, stream>>>(vl, vlT, NLIG, 64);
  transpose_f<<<dim3((NREC * 64 + 255) / 256), dim3(256), 0, stream>>>(vr, vrT, NREC, 64);

  // --- lig attends to rec ---
  score_wmma<<<dim3((NLIG + 15) / 16, (NREC + 63) / 64), dim3(32), 0, stream>>>(
      ql, kr, mask, (size_t)NREC, (size_t)1, P, NLIG, NREC);
  softmax_rows<<<dim3(NLIG), dim3(256), 0, stream>>>(P, NLIG, NREC);
  lin_hf(P, vrT, attL, NLIG, NREC, 64, NREC, stream);   // att = P @ vr
  layernorm64<<<dim3(NLIG), dim3(64), 0, stream>>>(attL, nullptr, g_lig, b_lig,
                                                   (float*)d_out, NLIG);

  // --- rec attends to lig (mask transposed via strides) ---
  score_wmma<<<dim3((NREC + 15) / 16, (NLIG + 63) / 64), dim3(32), 0, stream>>>(
      qr, kl, mask, (size_t)1, (size_t)NREC, P, NREC, NLIG);
  softmax_rows<<<dim3(NREC), dim3(256), 0, stream>>>(P, NREC, NLIG);
  lin_hf(P, vlT, attR, NREC, NLIG, 64, NLIG, stream);   // att = P @ vl
  layernorm64<<<dim3(NREC), dim3(64), 0, stream>>>(attR, nullptr, g_rec, b_rec,
                                                   (float*)d_out + (size_t)NLIG * 64, NREC);
}
